// GraphUnet_34385508172062
// MI455X (gfx1250) — matrix operand order repaired and checked by reading, not compiled, not run on hardware
//
#include <hip/hip_runtime.h>
#include <hip/hip_bf16.h>

// ---------------------------------------------------------------------------
// Types for CDNA5 WMMA / TDM
// ---------------------------------------------------------------------------
typedef __bf16 v16bf __attribute__((ext_vector_type(16)));
typedef float  v8f   __attribute__((ext_vector_type(8)));
typedef unsigned int v4u __attribute__((ext_vector_type(4)));
typedef int v4i  __attribute__((ext_vector_type(4)));
typedef int v8i_ __attribute__((ext_vector_type(8)));

// Problem constants (match reference)
#define NN    4096
#define DIMD  320
#define KK0   3686
#define KK1   2580
#define K0P   3712   // pad to multiple of 64
#define K1P   2624   // pad to multiple of 64

#if defined(__HIP_DEVICE_COMPILE__) && __has_builtin(__builtin_amdgcn_tensor_load_to_lds)
#define USE_TDM 1
#else
#define USE_TDM 0
#endif

#if __has_include(<hip/amd_detail/amd_gfx1250_TDM.h>)
#define TDM_6ARG 1
#else
#define TDM_6ARG 0
#endif

// ---------------------------------------------------------------------------
// Helpers
// ---------------------------------------------------------------------------
__device__ __forceinline__ unsigned short f2bf(float f) {
    unsigned int u = __float_as_uint(f);
    unsigned int r = u + 0x7FFFu + ((u >> 16) & 1u);   // round-to-nearest-even
    return (unsigned short)(r >> 16);
}

#if USE_TDM
// Issue one TDM 2D tile load: global (row-major, strideElems per row, 2-byte
// elements) -> LDS (rows packed contiguously, tile0 elems per row).
// D# bit packing per CDNA5 ISA ch.8 (group0: control/lds/global addr, type=2;
// group1: data_size=2B, tensor dims (huge => no OOB), tile dims, dim0 stride).
__device__ __forceinline__ void tdm_load_2d(unsigned ldsAddr, const void* gaddr,
                                            unsigned tile0, unsigned tile1,
                                            unsigned long long strideElems) {
    unsigned long long ga = (unsigned long long)(uintptr_t)gaddr;
    v4u g0;
    g0[0] = 1u;                                          // count=1, user mode
    g0[1] = ldsAddr;                                     // lds_addr
    g0[2] = (unsigned)(ga & 0xFFFFFFFFu);                // global_addr[31:0]
    g0[3] = (unsigned)((ga >> 32) & 0x01FFFFFFu) | (2u << 30);  // addr[56:32] | type=2
    const unsigned td0 = 0x7FFFFFFFu, td1 = 0x7FFFFFFFu; // no OOB clipping
    v8i_ g1;
    g1[0] = (int)(1u << 16);                             // data_size=1 -> 2 bytes
    g1[1] = (int)((td0 & 0xFFFFu) << 16);                // tensor_dim0 lo16
    g1[2] = (int)((td0 >> 16) | ((td1 & 0xFFFFu) << 16));// dim0 hi16 | dim1 lo16
    g1[3] = (int)((td1 >> 16) | (tile0 << 16));          // dim1 hi16 | tile_dim0
    g1[4] = (int)(tile1 & 0xFFFFu);                      // tile_dim1 | tile_dim2=0
    g1[5] = (int)(unsigned)(strideElems & 0xFFFFFFFFu);  // dim0_stride[31:0]
    g1[6] = (int)(unsigned)((strideElems >> 32) & 0xFFFFu); // dim0_stride[47:32]
    g1[7] = 0;
    v4i z; z[0] = 0; z[1] = 0; z[2] = 0; z[3] = 0;       // 2D: groups 2/3 unused
#if TDM_6ARG
    v8i_ z8; for (int i = 0; i < 8; ++i) z8[i] = 0;
    __builtin_amdgcn_tensor_load_to_lds(g0, g1, z, z, z8, 0);
#else
    __builtin_amdgcn_tensor_load_to_lds(g0, g1, z, z, 0);
#endif
}
#endif

// ---------------------------------------------------------------------------
// fp32 -> bf16 pack with zero padding (row-major dst).
// ---------------------------------------------------------------------------
__global__ void k_pack(unsigned short* __restrict__ dst, long ldd,
                       const float* __restrict__ src, long lds,
                       int rows, int cols, int rowsPad, int colsPad) {
    long i = (long)blockIdx.x * blockDim.x + threadIdx.x;
    long tot = (long)rowsPad * colsPad;
    if (i >= tot) return;
    int r = (int)(i / colsPad), c = (int)(i % colsPad);
    unsigned short v = 0;
    if (r < rows && c < cols) v = f2bf(src[(long)r * lds + c]);
    dst[(long)r * ldd + c] = v;
}

// fp32 -> bf16 transposed pack with zero padding.
// dst has dRowsPad x dColsPad (dst[rd][cd] = src[cd][rd]).
__global__ void k_pack_t(unsigned short* __restrict__ dst, long ldd,
                         const float* __restrict__ src, long lds,
                         int srcRows, int srcCols, int dRowsPad, int dColsPad) {
    long i = (long)blockIdx.x * blockDim.x + threadIdx.x;
    long tot = (long)dRowsPad * dColsPad;
    if (i >= tot) return;
    int rd = (int)(i / dColsPad), cd = (int)(i % dColsPad);
    unsigned short v = 0;
    if (cd < srcRows && rd < srcCols) v = f2bf(src[(long)cd * lds + rd]);
    dst[(long)rd * ldd + cd] = v;
}

// Sub-adjacency gather (bf16 -> bf16), zero padded.
__global__ void k_gather_A(unsigned short* __restrict__ dst, long ldd,
                           const unsigned short* __restrict__ src, long lds,
                           const int* __restrict__ idx, int k, int dimPad) {
    long i = (long)blockIdx.x * blockDim.x + threadIdx.x;
    long tot = (long)dimPad * dimPad;
    if (i >= tot) return;
    int r = (int)(i / dimPad), c = (int)(i % dimPad);
    unsigned short v = 0;
    if (r < k && c < k) v = src[(long)idx[r] * lds + idx[c]];
    dst[(long)r * ldd + c] = v;
}

// Pool gather + scale -> row-major (optional) and transposed (optional) bf16.
__global__ void k_gather_scale(unsigned short* __restrict__ dst, long ldd,
                               unsigned short* __restrict__ dstT, long ldT,
                               const float* __restrict__ src, long lds,
                               const int* __restrict__ idx,
                               const float* __restrict__ vals,
                               int k, int cols, int rowsPad) {
    long i = (long)blockIdx.x * blockDim.x + threadIdx.x;
    long tot = (long)rowsPad * cols;
    if (i >= tot) return;
    int r = (int)(i / cols), c = (int)(i % cols);
    unsigned short v = 0;
    if (r < k) v = f2bf(src[(long)idx[r] * lds + c] * vals[r]);
    if (dst)  dst[(long)r * ldd + c] = v;
    if (dstT) dstT[(long)c * ldT + r] = v;
}

// Unpool scatter into pre-zeroed TRANSPOSED bf16 buffer: dstT[c][idx[r]] = src[r][c]
__global__ void k_scatter_bf_t(unsigned short* __restrict__ dstT, long ldT,
                               const int* __restrict__ idx,
                               const float* __restrict__ src, long lds,
                               int k, int cols) {
    long i = (long)blockIdx.x * blockDim.x + threadIdx.x;
    long tot = (long)k * cols;
    if (i >= tot) return;
    int r = (int)(i / cols), c = (int)(i % cols);
    dstT[(long)c * ldT + idx[r]] = f2bf(src[(long)r * lds + c]);
}

__global__ void k_fill_u16(unsigned short* __restrict__ p, long n) {
    long i = (long)blockIdx.x * blockDim.x + threadIdx.x;
    if (i < n) p[i] = 0;
}

// Pool scores: s[i] = sigmoid((dot(X[i], pw) + pb) / 100)
__global__ void k_scores(const float* __restrict__ X, long ldx,
                         const float* __restrict__ pw, const float* __restrict__ pb,
                         float* __restrict__ s, int rows, int cols) {
    int i = blockIdx.x * blockDim.x + threadIdx.x;
    if (i >= rows) return;
    float acc = 0.f;
    const float* xr = X + (long)i * ldx;
    for (int c = 0; c < cols; ++c) acc += xr[c] * pw[c];
    float z = (acc + pb[0]) * 0.01f;
    s[i] = 1.0f / (1.0f + expf(-z));
}

// Exact top-k by rank counting (matches jax.lax.top_k ordering, ties -> lower idx)
__global__ void k_topk(const float* __restrict__ s, int n, int k,
                       int* __restrict__ idx, float* __restrict__ vals) {
    int i = blockIdx.x * blockDim.x + threadIdx.x;
    if (i >= n) return;
    float si = s[i];
    int rank = 0;
    for (int j = 0; j < n; ++j) {
        float sj = s[j];
        rank += (sj > si) || (sj == si && j < i);
    }
    if (rank < k) { idx[rank] = i; vals[rank] = si; }
}

__global__ void k_diag_one(float* __restrict__ p, int k, long ld) {
    int i = blockIdx.x * blockDim.x + threadIdx.x;
    if (i < k) p[(long)i * ld + i] = 1.0f;
}

// ---------------------------------------------------------------------------
// bf16 WMMA GEMM:  C[M,N](fp32) = A[M,K](bf16 row-major) * B[K,N]
// B is supplied TRANSPOSED: BT[n][k], leading dim ldbt.  K multiple of 64.
// Double-buffered pipeline: Tensor Data Mover fetches tile k+1 into the
// ping-pong LDS buffer while WMMAs consume tile k (s_wait_tensorcnt 1 =
// "next still in flight, current resident", per-wave TDM ordering).
// 64x64 output tile, 128 thr (4 wave32), staged BK=64 (2 WMMA K-steps/stage).
// Epilogue: (+bias[col]) -> relu -> (+res[row,col]); bounded store (Mv,Nv).
// ---------------------------------------------------------------------------
__global__ __launch_bounds__(128)
void k_gemm_bf16(const unsigned short* __restrict__ A, long lda,
                 const unsigned short* __restrict__ BT, long ldbt,
                 float* __restrict__ C, long ldc,
                 int Mv, int Nv, int K,
                 const float* __restrict__ bias,
                 const float* __restrict__ res, long ldres,
                 int relu) {
    __shared__ __align__(16) unsigned short As[2 * 64 * 64];  // ping-pong [m][k]
    __shared__ __align__(16) unsigned short Bs[2 * 64 * 64];  // ping-pong [n][k]

    const int t    = threadIdx.x;
    const int w    = t >> 5;          // wave 0..3 -> 16 output rows each
    const int lane = t & 31;
    const int half = lane >> 4;       // 0/1 K-half per ISA bf16 striping
    const int l16  = lane & 15;
    const long row0 = (long)blockIdx.y * 64;
    const long col0 = (long)blockIdx.x * 64;
    const int steps = K >> 6;

    v8f acc[4];
    #pragma unroll
    for (int i = 0; i < 4; ++i)
        #pragma unroll
        for (int j = 0; j < 8; ++j) acc[i][j] = 0.0f;

#if USE_TDM
    const unsigned asBase = (unsigned)(uintptr_t)(void*)As;
    const unsigned bsBase = (unsigned)(uintptr_t)(void*)Bs;
    // prologue: fetch stage 0 into buffer 0
    if (w == 0) tdm_load_2d(asBase, A + row0 * lda, 64, 64, (unsigned long long)lda);
    if (w == 1) tdm_load_2d(bsBase, BT + col0 * ldbt, 64, 64, (unsigned long long)ldbt);
#endif

    for (int it = 0; it < steps; ++it) {
        const int cur = it & 1;
        __syncthreads();   // all waves done reading buffer cur^1 (iteration it-1)
#if USE_TDM
        if (w < 2) {
            if (it + 1 < steps) {
                long kn = (long)(it + 1) << 6;
                if (w == 0)
                    tdm_load_2d(asBase + (cur ^ 1) * 8192u, A + row0 * lda + kn,
                                64, 64, (unsigned long long)lda);
                else
                    tdm_load_2d(bsBase + (cur ^ 1) * 8192u, BT + col0 * ldbt + kn,
                                64, 64, (unsigned long long)ldbt);
                __builtin_amdgcn_s_wait_tensorcnt(1);  // current tile resident
            } else {
                __builtin_amdgcn_s_wait_tensorcnt(0);  // last tile resident
            }
        }
#else
        {
            long k0 = (long)it << 6;
            #pragma unroll
            for (int ch = 0; ch < 4; ++ch) {
                int c  = t + ch * 128;
                int r  = c >> 3, c8 = (c & 7) << 3;
                uint4 va = *(const uint4*)(A  + (row0 + r) * lda  + k0 + c8);
                uint4 vb = *(const uint4*)(BT + (col0 + r) * ldbt + k0 + c8);
                *(uint4*)&As[cur * 4096 + r * 64 + c8] = va;
                *(uint4*)&Bs[cur * 4096 + r * 64 + c8] = vb;
            }
        }
#endif
        __syncthreads();

        const unsigned short* Ac = &As[cur * 4096];
        const unsigned short* Bc = &Bs[cur * 4096];
        #pragma unroll
        for (int j = 0; j < 2; ++j) {   // two 16x16x32 K-steps per stage
            // A fragment: lane holds row w*16+l16; K in {half*8..+7, +16..}
            union { uint4 q[2]; v16bf v; } ua;
            const uint4* ap = (const uint4*)&Ac[(w * 16 + l16) * 64 + j * 32 + half * 8];
            ua.q[0] = ap[0];
            ua.q[1] = ap[2];
            #pragma unroll
            for (int nt = 0; nt < 4; ++nt) {
                // B fragment: lane holds col nt*16+l16; K in {half*8..+7, +16..}
                union { uint4 q[2]; v16bf v; } ub;
                const uint4* bp =
                    (const uint4*)&Bc[(nt * 16 + l16) * 64 + j * 32 + half * 8];
                ub.q[0] = bp[0];
                ub.q[1] = bp[2];
                acc[nt] = __builtin_amdgcn_wmma_f32_16x16x32_bf16(
                    false, ua.v, false, ub.v, (short)0, acc[nt], false, false);
            }
        }
    }

    // epilogue (C/D layout: VGPR i -> M = i + half*8, lane l16 -> N)
    #pragma unroll
    for (int nt = 0; nt < 4; ++nt) {
        #pragma unroll
        for (int i = 0; i < 8; ++i) {
            long row = row0 + w * 16 + half * 8 + i;
            long col = col0 + nt * 16 + l16;
            if (row < Mv && col < Nv) {
                float v = acc[nt][i];
                if (bias) v += bias[col];
                if (relu) v = v > 0.f ? v : 0.f;
                if (res)  v += res[row * ldres + col];
                C[row * ldc + col] = v;
            }
        }
    }
}

// ---------------------------------------------------------------------------
// Host orchestration
// ---------------------------------------------------------------------------
extern "C" void kernel_launch(void* const* d_in, const int* in_sizes, int n_in,
                              void* d_out, int out_size, void* d_ws, size_t ws_size,
                              hipStream_t stream) {
    (void)in_sizes; (void)n_in; (void)out_size; (void)ws_size;

    const float* Af      = (const float*)d_in[0];
    const float* Xf      = (const float*)d_in[1];
    const float* w_start = (const float*)d_in[2];
    const float* b_start = (const float*)d_in[3];
    const float* w_down0 = (const float*)d_in[4];
    const float* b_down0 = (const float*)d_in[5];
    const float* w_down1 = (const float*)d_in[6];
    const float* b_down1 = (const float*)d_in[7];
    const float* p_w0    = (const float*)d_in[8];
    const float* p_b0    = (const float*)d_in[9];
    const float* p_w1    = (const float*)d_in[10];
    const float* p_b1    = (const float*)d_in[11];
    const float* w_bott  = (const float*)d_in[12];
    const float* b_bott  = (const float*)d_in[13];
    const float* w_up0   = (const float*)d_in[14];
    const float* b_up0   = (const float*)d_in[15];
    const float* w_up1   = (const float*)d_in[16];
    const float* b_up1   = (const float*)d_in[17];
    const float* w_end   = (const float*)d_in[18];
    const float* b_end   = (const float*)d_in[19];

    float* out_X    = (float*)d_out;                          // [N, DIM]
    float* out_X0   = (float*)d_out + (size_t)NN * DIMD;      // [N, DIM]
    float* out_pool = (float*)d_out + (size_t)2 * NN * DIMD;  // [K0, K0]

    // bump allocator over workspace
    char* ws = (char*)d_ws;
    size_t off = 0;
    auto alloc = [&](size_t bytes) -> void* {
        void* p = ws + off;
        off += (bytes + 255) & ~(size_t)255;
        return p;
    };

    unsigned short* Abf   = (unsigned short*)alloc((size_t)NN * NN * 2);
    unsigned short* A1bf  = (unsigned short*)alloc((size_t)K0P * K0P * 2);
    unsigned short* A2bf  = (unsigned short*)alloc((size_t)K1P * K1P * 2);
    unsigned short* XbfT  = (unsigned short*)alloc((size_t)DIMD * NN * 2);  // B of A@X (transposed)
    unsigned short* Tbf   = (unsigned short*)alloc((size_t)NN * DIMD * 2);  // A of (AX)@W (row-major)
    unsigned short* WsT   = (unsigned short*)alloc((size_t)DIMD * DIMD * 2);
    unsigned short* Wd0T  = (unsigned short*)alloc((size_t)DIMD * DIMD * 2);
    unsigned short* Wd1T  = (unsigned short*)alloc((size_t)DIMD * DIMD * 2);
    unsigned short* WbT   = (unsigned short*)alloc((size_t)DIMD * DIMD * 2);
    unsigned short* Wu0T  = (unsigned short*)alloc((size_t)DIMD * DIMD * 2);
    unsigned short* Wu1T  = (unsigned short*)alloc((size_t)DIMD * DIMD * 2);
    unsigned short* WeT   = (unsigned short*)alloc((size_t)DIMD * 2 * DIMD * 2);
    unsigned short* Xp0bf = (unsigned short*)alloc((size_t)K0P * DIMD * 2);  // pool A and pool BT
    unsigned short* Xp0T  = (unsigned short*)alloc((size_t)DIMD * K0P * 2);  // B of A1@Xp0
    unsigned short* Xp1T  = (unsigned short*)alloc((size_t)DIMD * K1P * 2);  // B of A2@Xp1
    unsigned short* XuT   = (unsigned short*)alloc((size_t)DIMD * K0P * 2);  // B of A1@Xu
    unsigned short* Xu2T  = (unsigned short*)alloc((size_t)DIMD * NN * 2);   // B of A@Xu2
    unsigned short* Ycat  = (unsigned short*)alloc((size_t)NN * 2 * DIMD * 2);
    float* Tf   = (float*)alloc((size_t)NN * DIMD * 4);   // A@X fp32 scratch
    float* AX0  = (float*)alloc((size_t)NN * DIMD * 4);   // reused for end concat
    float* Xd0  = (float*)alloc((size_t)NN * DIMD * 4);
    float* Xd1  = (float*)alloc((size_t)K0P * DIMD * 4);
    float* Xb   = (float*)alloc((size_t)K1P * DIMD * 4);
    float* Xu   = (float*)alloc((size_t)K0P * DIMD * 4);
    float* Xu2  = (float*)alloc((size_t)NN * DIMD * 4);
    float* s0   = (float*)alloc((size_t)NN * 4);
    float* s1   = (float*)alloc((size_t)KK0 * 4);
    float* v0   = (float*)alloc((size_t)KK0 * 4);
    float* v1   = (float*)alloc((size_t)KK1 * 4);
    int*   idx0 = (int*)alloc((size_t)KK0 * 4);
    int*   idx1 = (int*)alloc((size_t)KK1 * 4);

    auto cdiv = [](long a, long b) { return (int)((a + b - 1) / b); };
    auto flat = [&](long n) { return dim3((unsigned)cdiv(n, 256)); };
    auto gemm = [&](const unsigned short* A, long lda, const unsigned short* BT, long ldbt,
                    float* C, long ldc, int Mpad, int Npad, int K, int Mv, int Nv,
                    const float* bias, const float* res, long ldres, int relu) {
        dim3 g((unsigned)(Npad / 64), (unsigned)(Mpad / 64));
        k_gemm_bf16<<<g, 128, 0, stream>>>(A, lda, BT, ldbt, C, ldc, Mv, Nv, K,
                                           bias, res, ldres, relu);
    };

    // ---- pack adjacency (row-major A operand) and all B operands transposed ----
    k_pack<<<flat((long)NN * NN), 256, 0, stream>>>(Abf, NN, Af, NN, NN, NN, NN, NN);
    k_pack_t<<<flat((long)DIMD * NN), 256, 0, stream>>>(XbfT, NN, Xf, DIMD, NN, DIMD, DIMD, NN);
    k_pack_t<<<flat((long)DIMD * DIMD), 256, 0, stream>>>(WsT,  DIMD, w_start, DIMD, DIMD, DIMD, DIMD, DIMD);
    k_pack_t<<<flat((long)DIMD * DIMD), 256, 0, stream>>>(Wd0T, DIMD, w_down0, DIMD, DIMD, DIMD, DIMD, DIMD);
    k_pack_t<<<flat((long)DIMD * DIMD), 256, 0, stream>>>(Wd1T, DIMD, w_down1, DIMD, DIMD, DIMD, DIMD, DIMD);
    k_pack_t<<<flat((long)DIMD * DIMD), 256, 0, stream>>>(WbT,  DIMD, w_bott,  DIMD, DIMD, DIMD, DIMD, DIMD);
    k_pack_t<<<flat((long)DIMD * DIMD), 256, 0, stream>>>(Wu0T, DIMD, w_up0,   DIMD, DIMD, DIMD, DIMD, DIMD);
    k_pack_t<<<flat((long)DIMD * DIMD), 256, 0, stream>>>(Wu1T, DIMD, w_up1,   DIMD, DIMD, DIMD, DIMD, DIMD);
    k_pack_t<<<flat((long)DIMD * 2 * DIMD), 256, 0, stream>>>(WeT, 2 * DIMD, w_end, DIMD,
                                                              2 * DIMD, DIMD, DIMD, 2 * DIMD);

    // ---- start GCN: X0 = relu((A@X)@Ws + b); write straight into output slot 2 ----
    gemm(Abf, NN, XbfT, NN, Tf, DIMD, NN, DIMD, NN, NN, DIMD, nullptr, nullptr, 0, 0);
    k_pack<<<flat((long)NN * DIMD), 256, 0, stream>>>(Tbf, DIMD, Tf, DIMD, NN, DIMD, NN, DIMD);
    gemm(Tbf, DIMD, WsT, DIMD, out_X0, DIMD, NN, DIMD, DIMD, NN, DIMD, b_start, nullptr, 0, 1);

    // ---- down0: AX0 = A@X0 (kept for end concat); Xd0 = relu(AX0@Wd0+b) ----
    k_pack_t<<<flat((long)DIMD * NN), 256, 0, stream>>>(XbfT, NN, out_X0, DIMD, NN, DIMD, DIMD, NN);
    gemm(Abf, NN, XbfT, NN, AX0, DIMD, NN, DIMD, NN, NN, DIMD, nullptr, nullptr, 0, 0);
    k_pack<<<flat((long)NN * DIMD), 256, 0, stream>>>(Tbf, DIMD, AX0, DIMD, NN, DIMD, NN, DIMD);
    gemm(Tbf, DIMD, Wd0T, DIMD, Xd0, DIMD, NN, DIMD, DIMD, NN, DIMD, b_down0, nullptr, 0, 1);

    // ---- pool0 ----
    k_scores<<<flat(NN), 256, 0, stream>>>(Xd0, DIMD, p_w0, p_b0, s0, NN, DIMD);
    k_topk<<<flat(NN), 256, 0, stream>>>(s0, NN, KK0, idx0, v0);
    k_gather_A<<<flat((long)K0P * K0P), 256, 0, stream>>>(A1bf, K0P, Abf, NN, idx0, KK0, K0P);
    k_gather_scale<<<flat((long)K0P * DIMD), 256, 0, stream>>>(Xp0bf, DIMD, Xp0T, K0P,
                                                               Xd0, DIMD, idx0, v0, KK0, DIMD, K0P);

    // ---- down1: Xd1 = relu((A1@Xp0)@Wd1 + b) ----
    gemm(A1bf, K0P, Xp0T, K0P, Tf, DIMD, K0P, DIMD, K0P, K0P, DIMD, nullptr, nullptr, 0, 0);
    k_pack<<<flat((long)K0P * DIMD), 256, 0, stream>>>(Tbf, DIMD, Tf, DIMD, K0P, DIMD, K0P, DIMD);
    gemm(Tbf, DIMD, Wd1T, DIMD, Xd1, DIMD, K0P, DIMD, DIMD, K0P, DIMD, b_down1, nullptr, 0, 1);

    // ---- pool1 ----
    k_scores<<<flat(KK0), 256, 0, stream>>>(Xd1, DIMD, p_w1, p_b1, s1, KK0, DIMD);
    k_topk<<<flat(KK0), 256, 0, stream>>>(s1, KK0, KK1, idx1, v1);
    k_gather_A<<<flat((long)K1P * K1P), 256, 0, stream>>>(A2bf, K1P, A1bf, K0P, idx1, KK1, K1P);
    k_gather_scale<<<flat((long)K1P * DIMD), 256, 0, stream>>>(nullptr, 0, Xp1T, K1P,
                                                               Xd1, DIMD, idx1, v1, KK1, DIMD, K1P);

    // ---- bottom: Xb = relu((A2@Xp1)@Wb + b) ----
    gemm(A2bf, K1P, Xp1T, K1P, Tf, DIMD, K1P, DIMD, K1P, K1P, DIMD, nullptr, nullptr, 0, 0);
    k_pack<<<flat((long)K1P * DIMD), 256, 0, stream>>>(Tbf, DIMD, Tf, DIMD, K1P, DIMD, K1P, DIMD);
    gemm(Tbf, DIMD, WbT, DIMD, Xb, DIMD, K1P, DIMD, DIMD, K1P, DIMD, b_bott, nullptr, 0, 1);

    // ---- up0: scatter Xb to K0 rows (transposed), Xu = relu((A1@Xu)@Wu0+b)+Xd1 ----
    k_fill_u16<<<flat((long)DIMD * K0P), 256, 0, stream>>>(XuT, (long)DIMD * K0P);
    k_scatter_bf_t<<<flat((long)KK1 * DIMD), 256, 0, stream>>>(XuT, K0P, idx1, Xb, DIMD, KK1, DIMD);
    gemm(A1bf, K0P, XuT, K0P, Tf, DIMD, K0P, DIMD, K0P, K0P, DIMD, nullptr, nullptr, 0, 0);
    k_pack<<<flat((long)K0P * DIMD), 256, 0, stream>>>(Tbf, DIMD, Tf, DIMD, K0P, DIMD, K0P, DIMD);
    gemm(Tbf, DIMD, Wu0T, DIMD, Xu, DIMD, K0P, DIMD, DIMD, K0P, DIMD, b_up0, Xd1, DIMD, 1);

    // ---- up1: scatter Xu to N rows (transposed), Xu2 = relu((A@Xu2)@Wu1+b)+Xd0 ----
    k_fill_u16<<<flat((long)DIMD * NN), 256, 0, stream>>>(Xu2T, (long)DIMD * NN);
    k_scatter_bf_t<<<flat((long)KK0 * DIMD), 256, 0, stream>>>(Xu2T, NN, idx0, Xu, DIMD, KK0, DIMD);
    gemm(Abf, NN, Xu2T, NN, Tf, DIMD, NN, DIMD, NN, NN, DIMD, nullptr, nullptr, 0, 0);
    k_pack<<<flat((long)NN * DIMD), 256, 0, stream>>>(Tbf, DIMD, Tf, DIMD, NN, DIMD, NN, DIMD);
    gemm(Tbf, DIMD, Wu1T, DIMD, Xu2, DIMD, NN, DIMD, DIMD, NN, DIMD, b_up1, Xd0, DIMD, 1);

    // ---- end layer: A@[Xu2 | X0] = [A@Xu2 | AX0]; Xout = relu(Ycat@We + b_end) ----
    k_pack_t<<<flat((long)DIMD * NN), 256, 0, stream>>>(Xu2T, NN, Xu2, DIMD, NN, DIMD, DIMD, NN);
    gemm(Abf, NN, Xu2T, NN, Tf, DIMD, NN, DIMD, NN, NN, DIMD, nullptr, nullptr, 0, 0);
    k_pack<<<flat((long)NN * DIMD), 256, 0, stream>>>(Ycat,        2 * DIMD, Tf,  DIMD, NN, DIMD, NN, DIMD);
    k_pack<<<flat((long)NN * DIMD), 256, 0, stream>>>(Ycat + DIMD, 2 * DIMD, AX0, DIMD, NN, DIMD, NN, DIMD);
    gemm(Ycat, 2 * DIMD, WeT, 2 * DIMD, out_X, DIMD, NN, DIMD, 2 * DIMD, NN, DIMD, b_end, nullptr, 0, 1);

    // ---- pool_out = Xp0 @ Xp0^T (BT layout of Xp0^T is Xp0 itself), unit diag ----
    gemm(Xp0bf, DIMD, Xp0bf, DIMD, out_pool, KK0, K0P, K0P, DIMD, KK0, KK0, nullptr, nullptr, 0, 0);
    k_diag_one<<<flat(KK0), 256, 0, stream>>>(out_pool, KK0, KK0);
}
